// WindowAttention_8572754722836
// MI455X (gfx1250) — matrix operand order repaired and checked by reading, not compiled, not run on hardware
//
#include <hip/hip_runtime.h>
#include <hip/hip_bf16.h>

typedef __bf16 bf16_t;
typedef bf16_t v16bf __attribute__((ext_vector_type(16)));
typedef bf16_t v8bf  __attribute__((ext_vector_type(8)));
typedef bf16_t v4bf  __attribute__((ext_vector_type(4)));
typedef float  v8f   __attribute__((ext_vector_type(8)));

union ABFrag { v16bf v; v8bf h[2]; };

#define WMMA_BF16(a, b, c) \
  __builtin_amdgcn_wmma_f32_16x16x32_bf16(false, (a), false, (b), (short)0, (c), false, false)

__device__ __forceinline__ v8f zero8() {
  v8f z;
#pragma unroll
  for (int e = 0; e < 8; ++e) z[e] = 0.f;
  return z;
}

// round-to-nearest-even float -> bf16 (bit-exact, no reliance on __bf16 conv support)
__device__ __forceinline__ bf16_t f2bf(float f) {
  union { float f; unsigned u; } a; a.f = f;
  unsigned r = (a.u + 0x7FFFu + ((a.u >> 16) & 1u)) >> 16;
  union { unsigned short s; bf16_t b; } o; o.s = (unsigned short)r;
  return o.b;
}

// ---------------- problem constants ----------------
constexpr int BATCH = 16, HH = 56, WW = 56, DIM = 512, NHEAD = 16;
constexpr int WS = 7, WSQ = 49, HD = 32, NWIN = 64;   // 8x8 windows
constexpr int MTOK = BATCH * HH * WW;                  // 50176
constexpr int K3 = 3 * DIM;                            // 1536

// workspace layout (bytes)
constexpr size_t SZ_TOK_BF16 = (size_t)MTOK * DIM * 2;      // 51,380,224
constexpr size_t OFF_XS    = 0;                              // also aliased by attn-out
constexpr size_t OFF_WQKVT = SZ_TOK_BF16;                    // 1536*512*2
constexpr size_t OFF_WOUTT = OFF_WQKVT + (size_t)K3 * DIM * 2;
constexpr size_t OFF_Q     = OFF_WOUTT + (size_t)DIM * DIM * 2;
constexpr size_t OFF_K     = OFF_Q + SZ_TOK_BF16;
constexpr size_t OFF_V     = OFF_K + SZ_TOK_BF16;

// ---------------- kernel 1: cyclic shift (-3,-3) + fp32->bf16 ----------------
__global__ __launch_bounds__(256)
void k_shift_bf16(const float* __restrict__ x, bf16_t* __restrict__ xs) {
  int idx = blockIdx.x * 256 + threadIdx.x;           // each handles 4 elements
  int tok = idx >> 7;
  int c   = (idx & 127) << 2;
  if (tok >= MTOK) return;
  int bi = tok / 3136, rr = tok % 3136;
  int ys = rr / 56, xc = rr % 56;
  int ysrc = ys + 3; if (ysrc >= 56) ysrc -= 56;
  int xsrc = xc + 3; if (xsrc >= 56) xsrc -= 56;
  const float4 f = *(const float4*)(x + ((size_t)(bi * 56 + ysrc) * 56 + xsrc) * DIM + c);
  v4bf o;
  o[0] = f2bf(f.x); o[1] = f2bf(f.y); o[2] = f2bf(f.z); o[3] = f2bf(f.w);
  *(v4bf*)(xs + (size_t)tok * DIM + c) = o;
}

// ---------------- kernel 2/3: transpose + convert weights: dstT[n*K + k] = src[k*N + n]
__global__ __launch_bounds__(256)
void k_transpose_bf16(const float* __restrict__ src, bf16_t* __restrict__ dstT, int K, int N) {
  int idx = blockIdx.x * 256 + threadIdx.x;
  if (idx >= K * N) return;
  int n = idx / K, k2 = idx % K;
  dstT[idx] = f2bf(src[(size_t)k2 * N + n]);
}

// ---------------- kernel 4: QKV GEMM with windowed scatter epilogue ----------------
__global__ __launch_bounds__(256)
void k_gemm_qkv(const bf16_t* __restrict__ A, const bf16_t* __restrict__ BT,
                bf16_t* __restrict__ qd, bf16_t* __restrict__ kd, bf16_t* __restrict__ vd) {
  __shared__ bf16_t lA[128 * 32];
  __shared__ bf16_t lB[128 * 32];
  const int tid = threadIdx.x;
  const int lane = tid & 31, wv = tid >> 5;
  const int wvM = wv & 3, wvN = wv >> 2;               // 4 waves along M, 2 along N
  const int mlane = lane & 15, half = lane >> 4;
  const int mBlock = blockIdx.x * 128;
  const int nBlock = blockIdx.y * 128;

  v8f acc[2][4];
#pragma unroll
  for (int mt = 0; mt < 2; ++mt)
#pragma unroll
    for (int nt = 0; nt < 4; ++nt) acc[mt][nt] = zero8();

  for (int k0 = 0; k0 < DIM; k0 += 32) {
#pragma unroll
    for (int c = 0; c < 2; ++c) {
      int chunk = tid + c * 256;
      int row = chunk >> 2, c16 = chunk & 3;
      *(uint4*)&lA[row * 32 + c16 * 8] =
          *(const uint4*)(A + (size_t)(mBlock + row) * DIM + k0 + c16 * 8);
      *(uint4*)&lB[row * 32 + c16 * 8] =
          *(const uint4*)(BT + (size_t)(nBlock + row) * DIM + k0 + c16 * 8);
    }
    __syncthreads();
    ABFrag a[2], b[4];
#pragma unroll
    for (int mt = 0; mt < 2; ++mt) {
      int i = wvM * 32 + mt * 16 + mlane;
      a[mt].h[0] = *(const v8bf*)&lA[i * 32 + half * 8];
      a[mt].h[1] = *(const v8bf*)&lA[i * 32 + 16 + half * 8];
    }
#pragma unroll
    for (int nt = 0; nt < 4; ++nt) {
      int n = wvN * 64 + nt * 16 + mlane;
      b[nt].h[0] = *(const v8bf*)&lB[n * 32 + half * 8];
      b[nt].h[1] = *(const v8bf*)&lB[n * 32 + 16 + half * 8];
    }
#pragma unroll
    for (int mt = 0; mt < 2; ++mt)
#pragma unroll
      for (int nt = 0; nt < 4; ++nt)
        acc[mt][nt] = WMMA_BF16(a[mt].v, b[nt].v, acc[mt][nt]);
    __syncthreads();
  }

  // epilogue: scatter into windowed [b, head, win, 49, 32] bf16 q/k/v
#pragma unroll
  for (int mt = 0; mt < 2; ++mt) {
#pragma unroll
    for (int r = 0; r < 8; ++r) {
      int i = mBlock + wvM * 32 + mt * 16 + r + 8 * half;
      int bi = i / 3136, rr = i % 3136;
      int ys = rr / 56, xc = rr % 56;
      int wy = ys / 7, py = ys % 7;
      int wx = xc / 7, px = xc % 7;
      int win = wy * 8 + wx, p = py * 7 + px;
#pragma unroll
      for (int nt = 0; nt < 4; ++nt) {
        int j = nBlock + wvN * 64 + nt * 16 + mlane;
        int comp = j >> 9;            // 0=q 1=k 2=v
        int rem = j & 511;
        int h = rem >> 5, d = rem & 31;
        size_t dst = ((size_t)((bi * NHEAD + h) * NWIN + win) * WSQ + p) * HD + d;
        bf16_t val = f2bf(acc[mt][nt][r]);
        bf16_t* base = (comp == 0) ? qd : (comp == 1) ? kd : vd;
        base[dst] = val;
      }
    }
  }
}

// ---------------- kernel 5: shifted-window attention, one block per (b,h,window) --------
__global__ __launch_bounds__(128)
void k_attn(const bf16_t* __restrict__ qd, const bf16_t* __restrict__ kd,
            const bf16_t* __restrict__ vd, const float* __restrict__ pe,
            bf16_t* __restrict__ outd) {
  __shared__ bf16_t lq[64 * 32];
  __shared__ bf16_t lk[64 * 32];
  __shared__ bf16_t lvt[32 * 64];
  __shared__ bf16_t lp[64 * 64];
  __shared__ float lpe[169];

  const int tid = threadIdx.x, lane = tid & 31, wv = tid >> 5;
  const int mlane = lane & 15, half = lane >> 4;
  const int bid = blockIdx.x;
  const int win = bid & 63, h = (bid >> 6) & 15, bi = bid >> 10;
  const int wy = win >> 3, wx = win & 7;
  const size_t base = (size_t)((bi * NHEAD + h) * NWIN + win) * WSQ * HD;

  // load q,k tiles (49 rows, zero-pad to 64)
#pragma unroll
  for (int c = 0; c < 2; ++c) {
    int chunk = tid + c * 128;
    int row = chunk >> 2, c16 = chunk & 3;
    uint4 zq = {0, 0, 0, 0}, zk = {0, 0, 0, 0};
    if (row < WSQ) {
      zq = *(const uint4*)(qd + base + row * HD + c16 * 8);
      zk = *(const uint4*)(kd + base + row * HD + c16 * 8);
    }
    *(uint4*)&lq[row * 32 + c16 * 8] = zq;
    *(uint4*)&lk[row * 32 + c16 * 8] = zk;
  }
  // v transposed: lvt[d][tok]
#pragma unroll
  for (int c = 0; c < 16; ++c) {
    int idx = tid * 16 + c;                 // 0..2047
    int tok = idx >> 5, d = idx & 31;
    bf16_t v = (tok < WSQ) ? vd[base + tok * HD + d] : f2bf(0.f);
    lvt[d * 64 + tok] = v;
  }
  if (tid < 169) lpe[tid] = pe[tid];
  __syncthreads();

  // ---- stage 1: S = q k^T  (64x64 over 4 waves, 4 WMMA each) ----
  ABFrag aq;
  {
    int i = wv * 16 + mlane;
    aq.h[0] = *(const v8bf*)&lq[i * 32 + half * 8];
    aq.h[1] = *(const v8bf*)&lq[i * 32 + 16 + half * 8];
  }
  float pbuf[4][8];
#pragma unroll
  for (int nt = 0; nt < 4; ++nt) {
    ABFrag bk;
    int n = nt * 16 + mlane;
    bk.h[0] = *(const v8bf*)&lk[n * 32 + half * 8];
    bk.h[1] = *(const v8bf*)&lk[n * 32 + 16 + half * 8];
    v8f st = WMMA_BF16(aq.v, bk.v, zero8());
#pragma unroll
    for (int r = 0; r < 8; ++r) pbuf[nt][r] = st[r];
  }

  const bool maskY = (wy == 7), maskX = (wx == 7);
  const float scale = 0.17677669529663687f;   // 32^-0.5
#pragma unroll
  for (int nt = 0; nt < 4; ++nt) {
    int j = nt * 16 + mlane;
    int yj = j / 7, xj = j % 7;
#pragma unroll
    for (int r = 0; r < 8; ++r) {
      int i = wv * 16 + r + 8 * half;
      float s = pbuf[nt][r] * scale;
      if (j < WSQ && i < WSQ) {
        int yi = i / 7, xi = i % 7;
        s += lpe[(yj - yi + 6) * 13 + (xj - xi + 6)];
        if (maskY && ((yi >= 4) != (yj >= 4))) s = -1e30f;
        if (maskX && ((xi >= 4) != (xj >= 4))) s = -1e30f;
      } else {
        s = -1e30f;
      }
      pbuf[nt][r] = s;
    }
  }

  // ---- softmax across j (4 local values + 16-lane reduction) ----
#pragma unroll
  for (int r = 0; r < 8; ++r) {
    float m = fmaxf(fmaxf(pbuf[0][r], pbuf[1][r]), fmaxf(pbuf[2][r], pbuf[3][r]));
    for (int o = 1; o < 16; o <<= 1) m = fmaxf(m, __shfl_xor(m, o, 32));
    float sum = 0.f;
#pragma unroll
    for (int nt = 0; nt < 4; ++nt) {
      float e = __expf(pbuf[nt][r] - m);
      pbuf[nt][r] = e;
      sum += e;
    }
    for (int o = 1; o < 16; o <<= 1) sum += __shfl_xor(sum, o, 32);
    float inv = 1.f / sum;
    int i = wv * 16 + r + 8 * half;
#pragma unroll
    for (int nt = 0; nt < 4; ++nt) {
      int j = nt * 16 + mlane;
      lp[i * 64 + j] = f2bf(pbuf[nt][r] * inv);
    }
  }
  __syncthreads();

  // ---- stage 2: out = P @ v  (16 rows/wave, N=32, K=64) ----
  v8f co[2];
  co[0] = zero8(); co[1] = zero8();
#pragma unroll
  for (int ks = 0; ks < 2; ++ks) {
    ABFrag ap;
    int i = wv * 16 + mlane;
    ap.h[0] = *(const v8bf*)&lp[i * 64 + ks * 32 + half * 8];
    ap.h[1] = *(const v8bf*)&lp[i * 64 + ks * 32 + 16 + half * 8];
#pragma unroll
    for (int nt = 0; nt < 2; ++nt) {
      ABFrag bv;
      int d = nt * 16 + mlane;
      bv.h[0] = *(const v8bf*)&lvt[d * 64 + ks * 32 + half * 8];
      bv.h[1] = *(const v8bf*)&lvt[d * 64 + ks * 32 + 16 + half * 8];
      co[nt] = WMMA_BF16(ap.v, bv.v, co[nt]);
    }
  }
  // un-window store (still in shifted coordinates), bf16 [tok][head*32+d]
#pragma unroll
  for (int nt = 0; nt < 2; ++nt) {
#pragma unroll
    for (int r = 0; r < 8; ++r) {
      int i = wv * 16 + r + 8 * half;
      if (i < WSQ) {
        int y = wy * 7 + i / 7, x = wx * 7 + i % 7;
        size_t tok = (size_t)(bi * 56 + y) * 56 + x;
        outd[tok * DIM + h * HD + nt * 16 + mlane] = f2bf(co[nt][r]);
      }
    }
  }
}

// ---------------- kernel 6: output GEMM + bias, inverse roll fused into store ----------
__global__ __launch_bounds__(256)
void k_gemm_out(const bf16_t* __restrict__ A, const bf16_t* __restrict__ BT,
                const float* __restrict__ bias, float* __restrict__ out) {
  __shared__ bf16_t lA[128 * 32];
  __shared__ bf16_t lB[128 * 32];
  const int tid = threadIdx.x;
  const int lane = tid & 31, wv = tid >> 5;
  const int wvM = wv & 3, wvN = wv >> 2;
  const int mlane = lane & 15, half = lane >> 4;
  const int mBlock = blockIdx.x * 128;
  const int nBlock = blockIdx.y * 128;

  v8f acc[2][4];
#pragma unroll
  for (int mt = 0; mt < 2; ++mt)
#pragma unroll
    for (int nt = 0; nt < 4; ++nt) acc[mt][nt] = zero8();

  for (int k0 = 0; k0 < DIM; k0 += 32) {
#pragma unroll
    for (int c = 0; c < 2; ++c) {
      int chunk = tid + c * 256;
      int row = chunk >> 2, c16 = chunk & 3;
      *(uint4*)&lA[row * 32 + c16 * 8] =
          *(const uint4*)(A + (size_t)(mBlock + row) * DIM + k0 + c16 * 8);
      *(uint4*)&lB[row * 32 + c16 * 8] =
          *(const uint4*)(BT + (size_t)(nBlock + row) * DIM + k0 + c16 * 8);
    }
    __syncthreads();
    ABFrag a[2], b[4];
#pragma unroll
    for (int mt = 0; mt < 2; ++mt) {
      int i = wvM * 32 + mt * 16 + mlane;
      a[mt].h[0] = *(const v8bf*)&lA[i * 32 + half * 8];
      a[mt].h[1] = *(const v8bf*)&lA[i * 32 + 16 + half * 8];
    }
#pragma unroll
    for (int nt = 0; nt < 4; ++nt) {
      int n = wvN * 64 + nt * 16 + mlane;
      b[nt].h[0] = *(const v8bf*)&lB[n * 32 + half * 8];
      b[nt].h[1] = *(const v8bf*)&lB[n * 32 + 16 + half * 8];
    }
#pragma unroll
    for (int mt = 0; mt < 2; ++mt)
#pragma unroll
      for (int nt = 0; nt < 4; ++nt)
        acc[mt][nt] = WMMA_BF16(a[mt].v, b[nt].v, acc[mt][nt]);
    __syncthreads();
  }

#pragma unroll
  for (int mt = 0; mt < 2; ++mt) {
#pragma unroll
    for (int r = 0; r < 8; ++r) {
      int i = mBlock + wvM * 32 + mt * 16 + r + 8 * half;
      int bi = i / 3136, rr = i % 3136;
      int ys = rr / 56, xc = rr % 56;
      int y = ys + 3; if (y >= 56) y -= 56;     // roll back (+3,+3)
      int x = xc + 3; if (x >= 56) x -= 56;
      size_t rowoff = ((size_t)(bi * 56 + y) * 56 + x) * DIM;
#pragma unroll
      for (int nt = 0; nt < 4; ++nt) {
        int j = nBlock + wvN * 64 + nt * 16 + mlane;
        out[rowoff + j] = acc[mt][nt][r] + bias[j];
      }
    }
  }
}

// ---------------- launch ----------------
extern "C" void kernel_launch(void* const* d_in, const int* in_sizes, int n_in,
                              void* d_out, int out_size, void* d_ws, size_t ws_size,
                              hipStream_t stream) {
  const float* x     = (const float*)d_in[0];
  const float* w_qkv = (const float*)d_in[1];
  const float* pe    = (const float*)d_in[2];
  const float* w_out = (const float*)d_in[3];
  const float* b_out = (const float*)d_in[4];
  (void)in_sizes; (void)n_in; (void)out_size; (void)ws_size;

  char* ws = (char*)d_ws;
  bf16_t* xs    = (bf16_t*)(ws + OFF_XS);
  bf16_t* attnO = (bf16_t*)(ws + OFF_XS);     // alias: xs fully consumed before attn writes
  bf16_t* wqkvT = (bf16_t*)(ws + OFF_WQKVT);
  bf16_t* woutT = (bf16_t*)(ws + OFF_WOUTT);
  bf16_t* qd    = (bf16_t*)(ws + OFF_Q);
  bf16_t* kd    = (bf16_t*)(ws + OFF_K);
  bf16_t* vd    = (bf16_t*)(ws + OFF_V);
  float* out = (float*)d_out;

  k_shift_bf16<<<(MTOK * 128) / 256, 256, 0, stream>>>(x, xs);
  k_transpose_bf16<<<(K3 * DIM) / 256, 256, 0, stream>>>(w_qkv, wqkvT, DIM, K3);
  k_transpose_bf16<<<(DIM * DIM) / 256, 256, 0, stream>>>(w_out, woutT, DIM, DIM);
  k_gemm_qkv<<<dim3(MTOK / 128, K3 / 128), 256, 0, stream>>>(xs, wqkvT, qd, kd, vd);
  k_attn<<<BATCH * NHEAD * NWIN, 128, 0, stream>>>(qd, kd, vd, pe, attnO);
  k_gemm_out<<<dim3(MTOK / 128, DIM / 128), 256, 0, stream>>>(attnO, woutT, b_out, out);
}
